// GraniteSpeechConformerAttention_36438502539582
// MI455X (gfx1250) — matrix operand order repaired and checked by reading, not compiled, not run on hardware
//
#include <hip/hip_runtime.h>
#include <hip/hip_bf16.h>

typedef __bf16 bf16_t;
typedef __attribute__((ext_vector_type(16))) __bf16 v16bf;
typedef __attribute__((ext_vector_type(8)))  __bf16 v8bf;
typedef __attribute__((ext_vector_type(8)))  float  v8f;

// ---------------------------------------------------------------------------
// WMMA helpers (CDNA5 16x16x32 bf16, f32 accumulate)
// ---------------------------------------------------------------------------
__device__ inline v8f wmma_bf16(v16bf a, v16bf b, v8f c) {
  return __builtin_amdgcn_wmma_f32_16x16x32_bf16(false, a, false, b, (short)0, c,
                                                 false, false);
}

// A-fragment: 16x32 bf16, row-major source (global or LDS), stride lda.
// Lane layout (ISA 7.12.2): m = lane&15; half = lane>>4.
// elems 0..7  -> K = half*8 + e      (contiguous 16B)
// elems 8..15 -> K = 16 + half*8 + e (contiguous 16B)
__device__ inline v16bf load_afrag(const bf16_t* base, int lda, int m0, int k0,
                                   int lane) {
  int m = lane & 15, hf = lane >> 4;
  const bf16_t* p = base + (size_t)(m0 + m) * lda + k0 + hf * 8;
  v8bf lo = *(const v8bf*)p;
  v8bf hi = *(const v8bf*)(p + 16);
  return __builtin_shufflevector(lo, hi, 0, 1, 2, 3, 4, 5, 6, 7,
                                 8, 9, 10, 11, 12, 13, 14, 15);
}

// B-fragment where B = W^T and W is row-major (N x K), stride ldw.
// Lane n = lane&15 (column of B = row of W); half = lane>>4.
// elems 0..7 -> K = half*8 + e; elems 8..15 -> K = 16 + half*8 + e (2 x 16B).
__device__ inline v16bf load_bfrag_wt(const bf16_t* W, int ldw, int n0, int k0,
                                      int lane) {
  int n = lane & 15, hf = lane >> 4;
  const bf16_t* p = W + (size_t)(n0 + n) * ldw + k0 + hf * 8;
  v8bf lo = *(const v8bf*)p;
  v8bf hi = *(const v8bf*)(p + 16);
  return __builtin_shufflevector(lo, hi, 0, 1, 2, 3, 4, 5, 6, 7,
                                 8, 9, 10, 11, 12, 13, 14, 15);
}

// ---------------------------------------------------------------------------
// Kernel 1: LayerNorm + cast fp32 -> bf16, zero-pad rows n..npad-1
// ---------------------------------------------------------------------------
__global__ void ln_cast_kernel(const float* __restrict__ x,
                               const float* __restrict__ lw,
                               const float* __restrict__ lb,
                               bf16_t* __restrict__ xn,
                               int n, int npad, int d) {
  int row = blockIdx.x;  // over bs*npad
  int bb = row / npad, i = row % npad;
  bf16_t* out = xn + (size_t)row * d;
  if (i >= n) {
    for (int c = threadIdx.x; c < d; c += blockDim.x) out[c] = (bf16_t)0.0f;
    return;
  }
  const float* xr = x + ((size_t)bb * n + i) * d;
  float s = 0.f, s2 = 0.f;
  for (int c = threadIdx.x; c < d; c += blockDim.x) {
    float v = xr[c];
    s += v;
    s2 += v * v;
  }
#pragma unroll
  for (int o = 16; o > 0; o >>= 1) {
    s += __shfl_xor(s, o, 32);
    s2 += __shfl_xor(s2, o, 32);
  }
  __shared__ float rs[8], rs2[8];
  int wid = threadIdx.x >> 5, lid = threadIdx.x & 31;
  if (lid == 0) { rs[wid] = s; rs2[wid] = s2; }
  __syncthreads();
  s = 0.f; s2 = 0.f;
#pragma unroll
  for (int k = 0; k < 8; ++k) { s += rs[k]; s2 += rs2[k]; }
  float mu = s / (float)d;
  float var = s2 / (float)d - mu * mu;
  float inv = rsqrtf(var + 1e-5f);
  for (int c = threadIdx.x; c < d; c += blockDim.x)
    out[c] = (bf16_t)((xr[c] - mu) * inv * lw[c] + lb[c]);
}

// ---------------------------------------------------------------------------
// Kernel 2: elementwise fp32 -> bf16 cast (weights)
// ---------------------------------------------------------------------------
__global__ void cast_f32_bf16(const float* __restrict__ in,
                              bf16_t* __restrict__ out, int nelem) {
  int i = blockIdx.x * blockDim.x + threadIdx.x;
  if (i < nelem) out[i] = (bf16_t)in[i];
}

// ---------------------------------------------------------------------------
// Kernel 3: RelW (512 x 128) bf16: RelW[c][d] = rel[(257+c)*128 + d], c < 511;
// row 511 is zero.  W^T-form operand of P = Q @ Rel2 (contiguous B-frags).
// ---------------------------------------------------------------------------
__global__ void build_relw(const float* __restrict__ rel,
                           bf16_t* __restrict__ relw) {
  int idx = blockIdx.x * blockDim.x + threadIdx.x;
  if (idx >= 512 * 128) return;
  int c = idx >> 7;   // 0..511
  int dd = idx & 127; // 0..127
  float v = (c < 511) ? rel[(size_t)(257 + c) * 128 + dd] : 0.0f;
  relw[(size_t)c * 128 + dd] = (bf16_t)v;
}

// ---------------------------------------------------------------------------
// Kernel 4: LDS-staged, double-buffered GEMM  C = A(MxK) @ W^T   (bf16 out)
// Block: 512 threads (16 waves); block tile 128(M) x 256(N); K-stage 64.
// Wave tile 32x64 -> 8 C-tiles (16x16) = 64 acc VGPRs; staging 6 x v8bf.
// (Round 2's 32x128 wave tile spilled accumulators to scratch; this sizing
// keeps everything resident: ~130 VGPRs live.)
// LDS rows padded to 72 elements (144B, 16B-aligned) vs bank conflicts.
// Global loads for stage ks+1 issued before computing stage ks; stage ks+2
// prefetched (global_prefetch_b8).
// Columns >= splitN optionally scatter to per-(block,head) transposed V:
//   Vt[((blkb*8 + h)*128 + dh)*256 + j].
// ---------------------------------------------------------------------------
#define GTM 128
#define GTN 256
#define GTK 64
#define LDP 72       // padded LDS row stride (elements)
#define GTHREADS 512

__global__ void __launch_bounds__(GTHREADS, 1)
gemm_lds_bf16(const bf16_t* __restrict__ A,
              const bf16_t* __restrict__ W,
              bf16_t* __restrict__ C0,
              bf16_t* __restrict__ C1,
              int splitN, int K, int ldc0, int vtmode) {
  extern __shared__ char smem_raw[];
  bf16_t* la = (bf16_t*)smem_raw;                        // 2 x 128*72
  bf16_t* lw = (bf16_t*)(smem_raw + 2 * GTM * LDP * 2);  // 2 x 256*72
  int tid = threadIdx.x;
  int wave = tid >> 5, lane = tid & 31;
  int m0 = blockIdx.x * GTM;
  int n0 = blockIdx.y * GTN;
  int wm = (wave & 3) * 32;   // wave row offset inside block tile
  int wn = (wave >> 2) * 64;  // wave col offset inside block tile

  v8f acc[2][4];
#pragma unroll
  for (int a = 0; a < 2; ++a)
#pragma unroll
    for (int j = 0; j < 4; ++j) acc[a][j] = (v8f){};

  v8bf ra[2], rw[4];
  auto g_issue = [&](int k0) {
#pragma unroll
    for (int i = 0; i < 2; ++i) {  // A tile: 128x64 = 1024 16B-chunks
      int chunk = tid + GTHREADS * i;
      int row = chunk >> 3, off = (chunk & 7) * 8;
      ra[i] = *(const v8bf*)(A + (size_t)(m0 + row) * K + k0 + off);
    }
#pragma unroll
    for (int i = 0; i < 4; ++i) {  // W tile: 256x64 = 2048 16B-chunks
      int chunk = tid + GTHREADS * i;
      int row = chunk >> 3, off = (chunk & 7) * 8;
      rw[i] = *(const v8bf*)(W + (size_t)(n0 + row) * K + k0 + off);
    }
  };
  auto l_store = [&](int buf) {
    bf16_t* pa = la + buf * GTM * LDP;
    bf16_t* pw = lw + buf * GTN * LDP;
#pragma unroll
    for (int i = 0; i < 2; ++i) {
      int chunk = tid + GTHREADS * i;
      int row = chunk >> 3, off = (chunk & 7) * 8;
      *(v8bf*)(pa + row * LDP + off) = ra[i];
    }
#pragma unroll
    for (int i = 0; i < 4; ++i) {
      int chunk = tid + GTHREADS * i;
      int row = chunk >> 3, off = (chunk & 7) * 8;
      *(v8bf*)(pw + row * LDP + off) = rw[i];
    }
  };

  g_issue(0);
  l_store(0);
  __syncthreads();
  int nk = K / GTK;
  for (int ks = 0; ks < nk; ++ks) {
    if (ks + 1 < nk) g_issue((ks + 1) * GTK);
    if (ks + 2 < nk) {
      int k2 = (ks + 2) * GTK;
      __builtin_prefetch(A + (size_t)(m0 + (tid >> 2)) * K + k2, 0, 1);
      __builtin_prefetch(W + (size_t)(n0 + (tid >> 1)) * K + k2, 0, 1);
    }
    const bf16_t* pa = la + (ks & 1) * GTM * LDP;
    const bf16_t* pw = lw + (ks & 1) * GTN * LDP;
#pragma unroll
    for (int kk = 0; kk < GTK; kk += 32) {
      v16bf a0 = load_afrag(pa, LDP, wm, kk, lane);
      v16bf a1 = load_afrag(pa, LDP, wm + 16, kk, lane);
#pragma unroll
      for (int j = 0; j < 4; ++j) {
        v16bf bf = load_bfrag_wt(pw, LDP, wn + j * 16, kk, lane);
        acc[0][j] = wmma_bf16(a0, bf, acc[0][j]);
        acc[1][j] = wmma_bf16(a1, bf, acc[1][j]);
      }
    }
    __syncthreads();
    if (ks + 1 < nk) {
      l_store((ks + 1) & 1);
      __syncthreads();
    }
  }

  int cn = lane & 15, ch = lane >> 4;
#pragma unroll
  for (int a = 0; a < 2; ++a)
#pragma unroll
    for (int j = 0; j < 4; ++j) {
      int col = n0 + wn + j * 16 + cn;
#pragma unroll
      for (int r = 0; r < 8; ++r) {
        int row = m0 + wm + a * 16 + ch * 8 + r;
        bf16_t v = (bf16_t)acc[a][j][r];
        if (col < splitN) {
          C0[(size_t)row * ldc0 + col] = v;
        } else if (vtmode) {
          int c = col - splitN;  // 0..1023
          int hh = c >> 7, dh = c & 127;
          int blkb = row >> 8, jj = row & 255;
          C1[(((size_t)blkb * 8 + hh) * 128 + dh) * 256 + jj] = v;
        } else {
          C1[(size_t)row * ldc0 + (col - splitN)] = v;
        }
      }
    }
}

// ---------------------------------------------------------------------------
// Kernel 5: block-local attention.  One workgroup per (b, blk, head);
// 8 waves x 32 rows.  All B-fragments are contiguous W^T-form loads.
// ---------------------------------------------------------------------------
__global__ void attn_kernel(const bf16_t* __restrict__ Qb,
                            const bf16_t* __restrict__ Kb,
                            const bf16_t* __restrict__ Vt,
                            const bf16_t* __restrict__ relw,
                            bf16_t* __restrict__ Ob,
                            int nb, int nr, float scale) {
  extern __shared__ float smem[];
  const int ld = 1024;   // h*dh
  int bid = blockIdx.x;  // bs*nb*h
  int head = bid & 7;
  int t = bid >> 3;
  int blk = t % nb, bb = t / nb;
  int wave = threadIdx.x >> 5, lane = threadIdx.x & 31;
  float* sstrip = smem + wave * (16 * 256);

  int blkb = bb * nb + blk;
  size_t rowbase = (size_t)blkb * 256;
  const bf16_t* Qbase = Qb + rowbase * ld + head * 128;
  const bf16_t* Kbase = Kb + rowbase * ld + head * 128;
  const bf16_t* Vbase = Vt + ((size_t)blkb * 8 + head) * 128 * 256;  // (dh x j)
  bf16_t* Obase = Ob + rowbase * ld + head * 128;
  bool lastblk = (blk == nb - 1) && (nr > 0);

  for (int rt = 0; rt < 2; ++rt) {
    int i0 = wave * 32 + rt * 16;

    v16bf qf[4];
#pragma unroll
    for (int ks = 0; ks < 4; ++ks)
      qf[ks] = load_afrag(Qbase, ld, i0, ks * 32, lane);

    // ---- S = Q K^T  (16 x 256) ----
    for (int nt = 0; nt < 16; ++nt) {
      v8f acc = (v8f){};
#pragma unroll
      for (int ks = 0; ks < 4; ++ks) {
        v16bf bf = load_bfrag_wt(Kbase, ld, nt * 16, ks * 32, lane);
        acc = wmma_bf16(qf[ks], bf, acc);
      }
      int cn = lane & 15, ch = lane >> 4;
#pragma unroll
      for (int r = 0; r < 8; ++r)
        sstrip[(ch * 8 + r) * 256 + nt * 16 + cn] = acc[r];
    }

    // ---- P = Q @ RelW^T (16 x 512), scatter-add into S band ----
    for (int ct = 0; ct < 32; ++ct) {
      v8f acc = (v8f){};
#pragma unroll
      for (int ks = 0; ks < 4; ++ks) {
        v16bf bf = load_bfrag_wt(relw, 128, ct * 16, ks * 32, lane);
        acc = wmma_bf16(qf[ks], bf, acc);
      }
      int cn = lane & 15, ch = lane >> 4;
#pragma unroll
      for (int r = 0; r < 8; ++r) {
        int li = ch * 8 + r;
        int i = i0 + li;
        int c = ct * 16 + cn;
        int j = i - c + 255;
        if (j >= 0 && j < 256) sstrip[li * 256 + j] += acc[r];
      }
    }

    // ---- softmax over 256 columns, per row ----
    for (int r = 0; r < 16; ++r) {
      int i = i0 + r;
      float v[8];
      float mx = -3.0e38f;
#pragma unroll
      for (int q = 0; q < 8; ++q) {
        int j = q * 32 + lane;
        float s = sstrip[r * 256 + j] * scale;
        if (lastblk && (j >= nr || i >= nr)) s = -3.0e38f;
        v[q] = s;
        mx = fmaxf(mx, s);
      }
#pragma unroll
      for (int o = 16; o > 0; o >>= 1) mx = fmaxf(mx, __shfl_xor(mx, o, 32));
      float sum = 0.f;
#pragma unroll
      for (int q = 0; q < 8; ++q) {
        float e = __expf(v[q] - mx);
        v[q] = e;
        sum += e;
      }
#pragma unroll
      for (int o = 16; o > 0; o >>= 1) sum += __shfl_xor(sum, o, 32);
      float inv = 1.0f / sum;
#pragma unroll
      for (int q = 0; q < 8; ++q) sstrip[r * 256 + q * 32 + lane] = v[q] * inv;
    }

    // ---- O = attn @ V  (16 x 128), B from transposed Vt (contiguous) ----
    {
      int m = lane & 15, ah = lane >> 4;
      int cn = lane & 15, ch = lane >> 4;
      for (int nt = 0; nt < 8; ++nt) {
        v8f acc = (v8f){};
#pragma unroll
        for (int ks = 0; ks < 8; ++ks) {
          v16bf af;
#pragma unroll
          for (int e = 0; e < 16; ++e) {
            int kk = (e < 8) ? (ah * 8 + e) : (16 + ah * 8 + (e - 8));
            af[e] = (bf16_t)sstrip[m * 256 + ks * 32 + kk];
          }
          v16bf bf = load_bfrag_wt(Vbase, 256, nt * 16, ks * 32, lane);
          acc = wmma_bf16(af, bf, acc);
        }
#pragma unroll
        for (int r = 0; r < 8; ++r)
          Obase[(size_t)(i0 + ch * 8 + r) * ld + nt * 16 + cn] = (bf16_t)acc[r];
      }
    }
  }
}

// ---------------------------------------------------------------------------
// Kernel 6: LDS-staged out-projection: out = O @ Wo^T + bo  (fp32 out,
// padded rows i >= n dropped).  Same tiling as gemm_lds_bf16.
// ---------------------------------------------------------------------------
__global__ void __launch_bounds__(GTHREADS, 1)
gemm_out_lds(const bf16_t* __restrict__ A,
             const bf16_t* __restrict__ W,
             const float* __restrict__ bias,
             float* __restrict__ out,
             int npad, int n, int N, int K) {
  extern __shared__ char smem_raw[];
  bf16_t* la = (bf16_t*)smem_raw;
  bf16_t* lw = (bf16_t*)(smem_raw + 2 * GTM * LDP * 2);
  int tid = threadIdx.x;
  int wave = tid >> 5, lane = tid & 31;
  int m0 = blockIdx.x * GTM;
  int n0 = blockIdx.y * GTN;
  int wm = (wave & 3) * 32;
  int wn = (wave >> 2) * 64;

  v8f acc[2][4];
#pragma unroll
  for (int a = 0; a < 2; ++a)
#pragma unroll
    for (int j = 0; j < 4; ++j) acc[a][j] = (v8f){};

  v8bf ra[2], rw[4];
  auto g_issue = [&](int k0) {
#pragma unroll
    for (int i = 0; i < 2; ++i) {
      int chunk = tid + GTHREADS * i;
      int row = chunk >> 3, off = (chunk & 7) * 8;
      ra[i] = *(const v8bf*)(A + (size_t)(m0 + row) * K + k0 + off);
    }
#pragma unroll
    for (int i = 0; i < 4; ++i) {
      int chunk = tid + GTHREADS * i;
      int row = chunk >> 3, off = (chunk & 7) * 8;
      rw[i] = *(const v8bf*)(W + (size_t)(n0 + row) * K + k0 + off);
    }
  };
  auto l_store = [&](int buf) {
    bf16_t* pa = la + buf * GTM * LDP;
    bf16_t* pw = lw + buf * GTN * LDP;
#pragma unroll
    for (int i = 0; i < 2; ++i) {
      int chunk = tid + GTHREADS * i;
      int row = chunk >> 3, off = (chunk & 7) * 8;
      *(v8bf*)(pa + row * LDP + off) = ra[i];
    }
#pragma unroll
    for (int i = 0; i < 4; ++i) {
      int chunk = tid + GTHREADS * i;
      int row = chunk >> 3, off = (chunk & 7) * 8;
      *(v8bf*)(pw + row * LDP + off) = rw[i];
    }
  };

  g_issue(0);
  l_store(0);
  __syncthreads();
  int nk = K / GTK;
  for (int ks = 0; ks < nk; ++ks) {
    if (ks + 1 < nk) g_issue((ks + 1) * GTK);
    if (ks + 2 < nk) {
      int k2 = (ks + 2) * GTK;
      __builtin_prefetch(A + (size_t)(m0 + (tid >> 2)) * K + k2, 0, 1);
      __builtin_prefetch(W + (size_t)(n0 + (tid >> 1)) * K + k2, 0, 1);
    }
    const bf16_t* pa = la + (ks & 1) * GTM * LDP;
    const bf16_t* pw = lw + (ks & 1) * GTN * LDP;
#pragma unroll
    for (int kk = 0; kk < GTK; kk += 32) {
      v16bf a0 = load_afrag(pa, LDP, wm, kk, lane);
      v16bf a1 = load_afrag(pa, LDP, wm + 16, kk, lane);
#pragma unroll
      for (int j = 0; j < 4; ++j) {
        v16bf bf = load_bfrag_wt(pw, LDP, wn + j * 16, kk, lane);
        acc[0][j] = wmma_bf16(a0, bf, acc[0][j]);
        acc[1][j] = wmma_bf16(a1, bf, acc[1][j]);
      }
    }
    __syncthreads();
    if (ks + 1 < nk) {
      l_store((ks + 1) & 1);
      __syncthreads();
    }
  }

  int cn = lane & 15, ch = lane >> 4;
#pragma unroll
  for (int a = 0; a < 2; ++a)
#pragma unroll
    for (int j = 0; j < 4; ++j) {
      int col = n0 + wn + j * 16 + cn;
      float bv = bias[col];
#pragma unroll
      for (int r = 0; r < 8; ++r) {
        int rowp = m0 + wm + a * 16 + ch * 8 + r;
        int bb = rowp / npad, i = rowp % npad;
        if (i < n) out[((size_t)bb * n + i) * N + col] = acc[a][j][r] + bv;
      }
    }
}

// ---------------------------------------------------------------------------
// Host launcher
// ---------------------------------------------------------------------------
extern "C" void kernel_launch(void* const* d_in, const int* in_sizes, int n_in,
                              void* d_out, int out_size, void* d_ws,
                              size_t ws_size, hipStream_t stream) {
  (void)in_sizes; (void)n_in; (void)out_size; (void)ws_size;
  const float* x    = (const float*)d_in[0];
  const float* ln_w = (const float*)d_in[1];
  const float* ln_b = (const float*)d_in[2];
  const float* Wq   = (const float*)d_in[3];
  const float* Wkv  = (const float*)d_in[4];
  const float* Wo   = (const float*)d_in[5];
  const float* bo   = (const float*)d_in[6];
  const float* rel  = (const float*)d_in[7];
  // d_in[8] = context_size (256, matches setup_inputs; fixed here)

  const int bs = 16, n = 4000, d = 1024, h = 8, cs = 256;
  const int nb = (n + cs - 1) / cs;   // 16
  const int npad = nb * cs;           // 4096
  const int nr = n % cs;              // 160
  const int M = bs * npad;            // 65536
  const float scale = 0.08838834764831845f;  // 128^-0.5

  size_t off = 0;
  auto bump = [&](size_t bytes) {
    char* p = (char*)d_ws + off;
    off += (bytes + 255) & ~(size_t)255;
    return p;
  };
  const size_t actBytes = (size_t)M * d * sizeof(bf16_t);
  bf16_t* Xn   = (bf16_t*)bump(actBytes);
  bf16_t* Qb   = (bf16_t*)bump(actBytes);
  bf16_t* Kb   = (bf16_t*)bump(actBytes);
  bf16_t* Vt   = (bf16_t*)bump(actBytes);   // per-(block,head) transposed V
  bf16_t* Ob   = (bf16_t*)bump(actBytes);
  bf16_t* WqB  = (bf16_t*)bump((size_t)1024 * 1024 * 2);
  bf16_t* WkvB = (bf16_t*)bump((size_t)2 * 1024 * 1024 * 2);
  bf16_t* WoB  = (bf16_t*)bump((size_t)1024 * 1024 * 2);
  bf16_t* RelW = (bf16_t*)bump((size_t)512 * 128 * 2);

  const size_t gemmLds = 2 * (size_t)(GTM + GTN) * LDP * sizeof(bf16_t); // 110,592B
  const size_t attnLds = 8 * 16 * 256 * sizeof(float);                   // 131,072B

  // 1. LayerNorm + cast (+ zero padding rows)
  ln_cast_kernel<<<M, 256, 0, stream>>>(x, ln_w, ln_b, Xn, n, npad, d);

  // 2. Weight casts + RelW
  cast_f32_bf16<<<(1024 * 1024 + 255) / 256, 256, 0, stream>>>(Wq, WqB, 1024 * 1024);
  cast_f32_bf16<<<(2 * 1024 * 1024 + 255) / 256, 256, 0, stream>>>(Wkv, WkvB, 2 * 1024 * 1024);
  cast_f32_bf16<<<(1024 * 1024 + 255) / 256, 256, 0, stream>>>(Wo, WoB, 1024 * 1024);
  build_relw<<<(512 * 128 + 255) / 256, 256, 0, stream>>>(rel, RelW);

  // 3. Q = Xn Wq^T ; [K|V] = Xn Wkv^T (V written transposed per block-head)
  dim3 gq(M / GTM, 1024 / GTN);
  gemm_lds_bf16<<<gq, GTHREADS, gemmLds, stream>>>(Xn, WqB, Qb, Qb, 1024, 1024,
                                                   1024, 0);
  dim3 gkv(M / GTM, 2048 / GTN);
  gemm_lds_bf16<<<gkv, GTHREADS, gemmLds, stream>>>(Xn, WkvB, Kb, Vt, 1024, 1024,
                                                    1024, 1);

  // 4. Block attention: one workgroup per (b, blk, head)
  attn_kernel<<<bs * nb * h, 256, attnLds, stream>>>(Qb, Kb, Vt, RelW, Ob, nb, nr,
                                                     scale);

  // 5. out = O Wo^T + bo (fp32, padded rows dropped)
  dim3 go(M / GTM, 1024 / GTN);
  gemm_out_lds<<<go, GTHREADS, gemmLds, stream>>>(Ob, WoB, bo, (float*)d_out,
                                                  npad, n, 1024, 1024);
}
